// FastTemporalCrosscoder_82411832476229
// MI455X (gfx1250) — matrix operand order repaired and checked by reading, not compiled, not run on hardware
//
#include <hip/hip_runtime.h>
#include <hip/hip_bf16.h>
#include <stdint.h>

#define B_ROWS 2048
#define KDIM   3072      // T * d_in = 4 * 768
#define S_DIM  16384
#define TOPK   128
#define KSTEPS (KDIM / 32)

typedef __attribute__((ext_vector_type(16))) __bf16 v16bf;
typedef __attribute__((ext_vector_type(8)))  __bf16 v8bf;
typedef __attribute__((ext_vector_type(4)))  __bf16 v4bf;
typedef __attribute__((ext_vector_type(2)))  __bf16 v2bf;
typedef __attribute__((ext_vector_type(8)))  float  v8f;
typedef __attribute__((ext_vector_type(4)))  float  f32x4;
typedef __attribute__((ext_vector_type(2)))  float  f32x2;

__device__ __forceinline__ unsigned fkey(float f) {
  // order-preserving map: larger float -> larger uint
  unsigned u = __float_as_uint(f);
  return (u & 0x80000000u) ? ~u : (u | 0x80000000u);
}

// ---------------------------------------------------------------------------
// Kernel 1: pre = x(2048x3072) * W_enc(3072x16384) + b_enc  via bf16 WMMA.
// 256 threads = 8 waves (2x4 grid), 128x128 tile, BK=32, double-buffered LDS
// (register prefetch of tile i+1 overlaps the 8 WMMAs of tile i).
// f32->bf16 via __builtin_convertvector -> packed v_cvt (RNE).
// ---------------------------------------------------------------------------
__global__ __launch_bounds__(256) void enc_kernel(const float* __restrict__ X,
                                                  const float* __restrict__ W,
                                                  const float* __restrict__ benc,
                                                  float* __restrict__ pre) {
  // [buf][row][k] for A, [buf][n][k] (K-transposed) for B; stride 40 halves
  // (80B) keeps b128 fragment chunks 16B-aligned and spreads banks.
  __shared__ unsigned short lsA[2][128 * 40];   // 2 x 10KB
  __shared__ unsigned short lsB[2][128 * 40];   // 2 x 10KB

  const int tid  = threadIdx.x;
  const int lane = tid & 31;
  const int wave = tid >> 5;
  const int wm   = wave & 1;      // 2 waves down M
  const int wn   = wave >> 1;     // 4 waves across N
  const int lrow = lane & 15;
  const int lhi  = lane >> 4;
  const int mBase = blockIdx.x * 128;
  const int nBase = blockIdx.y * 128;

  // staging decomposition (per thread, per K-step):
  //  A: 4 float4 loads: row = (tid>>3)+32j, colgroup cg = tid&7 (4 cols each)
  //  B: 2 k-pairs: ngroup ng = tid&31 (4 cols), kp = (tid>>5)+8j, rows k=2kp,2kp+1
  const int aCg  = tid & 7;
  const int aR0  = tid >> 3;
  const int bNg  = tid & 31;
  const int bKp0 = tid >> 5;

  float4 ra[4];
  float4 rbl[2], rbh[2];

  auto load_tile = [&](int k0) {
#pragma unroll
    for (int j = 0; j < 4; ++j) {
      int row = aR0 + 32 * j;
      ra[j] = *(const float4*)&X[(size_t)(mBase + row) * KDIM + k0 + aCg * 4];
    }
#pragma unroll
    for (int j = 0; j < 2; ++j) {
      int kp = bKp0 + 8 * j;
      rbl[j] = *(const float4*)&W[(size_t)(k0 + 2 * kp) * S_DIM + nBase + bNg * 4];
      rbh[j] = *(const float4*)&W[(size_t)(k0 + 2 * kp + 1) * S_DIM + nBase + bNg * 4];
    }
  };

  auto store_tile = [&](int s) {
#pragma unroll
    for (int j = 0; j < 4; ++j) {
      int row = aR0 + 32 * j;
      f32x4 f = {ra[j].x, ra[j].y, ra[j].z, ra[j].w};
      *(v4bf*)&lsA[s][row * 40 + aCg * 4] =
          __builtin_convertvector(f, v4bf);                 // packed cvt + ds_store_b64
    }
#pragma unroll
    for (int j = 0; j < 2; ++j) {
      int kp = bKp0 + 8 * j;
      // pack (k=2kp, k=2kp+1) pair per column n -> one b32 per column
      f32x2 p0 = {rbl[j].x, rbh[j].x};
      f32x2 p1 = {rbl[j].y, rbh[j].y};
      f32x2 p2 = {rbl[j].z, rbh[j].z};
      f32x2 p3 = {rbl[j].w, rbh[j].w};
      *(v2bf*)&lsB[s][(bNg * 4 + 0) * 40 + 2 * kp] = __builtin_convertvector(p0, v2bf);
      *(v2bf*)&lsB[s][(bNg * 4 + 1) * 40 + 2 * kp] = __builtin_convertvector(p1, v2bf);
      *(v2bf*)&lsB[s][(bNg * 4 + 2) * 40 + 2 * kp] = __builtin_convertvector(p2, v2bf);
      *(v2bf*)&lsB[s][(bNg * 4 + 3) * 40 + 2 * kp] = __builtin_convertvector(p3, v2bf);
    }
  };

  v8f acc[4][2];
#pragma unroll
  for (int mt = 0; mt < 4; ++mt)
#pragma unroll
    for (int nt = 0; nt < 2; ++nt) acc[mt][nt] = (v8f)0.0f;

  load_tile(0);
  store_tile(0);
  __syncthreads();

  for (int it = 0; it < KSTEPS; ++it) {
    const int cur = it & 1;
    if (it + 1 < KSTEPS) load_tile((it + 1) * 32);   // prefetch into registers

    const unsigned short* As = lsA[cur];
    const unsigned short* Bs = lsB[cur];

    // A frag (16x32 bf16): lane L: M=L&15; K = (L>>4)*8 + [0..7], then +16
    v16bf af[4];
#pragma unroll
    for (int mt = 0; mt < 4; ++mt) {
      int base = (wm * 64 + mt * 16 + lrow) * 40 + lhi * 8;
      v8bf lo = *(const v8bf*)&As[base];
      v8bf hi = *(const v8bf*)&As[base + 16];
#pragma unroll
      for (int i = 0; i < 8; ++i) { af[mt][i] = lo[i]; af[mt][8 + i] = hi[i]; }
    }
    // B frag (32x16 bf16): lane L: N=L&15; K = (L>>4)*8 + [0..7], then +16
    v16bf bfr[2];
#pragma unroll
    for (int nt = 0; nt < 2; ++nt) {
      int base = (wn * 32 + nt * 16 + lrow) * 40 + lhi * 8;
      v8bf lo = *(const v8bf*)&Bs[base];
      v8bf hi = *(const v8bf*)&Bs[base + 16];
#pragma unroll
      for (int i = 0; i < 8; ++i) { bfr[nt][i] = lo[i]; bfr[nt][8 + i] = hi[i]; }
    }

#pragma unroll
    for (int mt = 0; mt < 4; ++mt)
#pragma unroll
      for (int nt = 0; nt < 2; ++nt)
        acc[mt][nt] = __builtin_amdgcn_wmma_f32_16x16x32_bf16(
            false, af[mt], false, bfr[nt], (short)0, acc[mt][nt], false, false);

    if (it + 1 < KSTEPS) store_tile(cur ^ 1);  // writers of buf^1 vs readers of buf: no conflict
    __syncthreads();
  }

  // epilogue: + b_enc, store pre. C layout: VGPR r -> row r + 8*(lane>=16), col lane&15
  float be[2];
#pragma unroll
  for (int nt = 0; nt < 2; ++nt) be[nt] = benc[nBase + wn * 32 + nt * 16 + lrow];
#pragma unroll
  for (int mt = 0; mt < 4; ++mt)
#pragma unroll
    for (int nt = 0; nt < 2; ++nt) {
      int col = nBase + wn * 32 + nt * 16 + lrow;
#pragma unroll
      for (int r = 0; r < 8; ++r) {
        int row = mBase + wm * 64 + mt * 16 + r + lhi * 8;
        pre[(size_t)row * S_DIM + col] = acc[mt][nt][r] + be[nt];
      }
    }
}

// ---------------------------------------------------------------------------
// Kernel 2: per-row exact top-128 via 4-pass radix select, relu+scatter in
// place, deterministic index-ordered compact (val,idx) list for decode.
// ---------------------------------------------------------------------------
__global__ __launch_bounds__(256) void topk_kernel(float* __restrict__ z,
                                                   float* __restrict__ tkv,
                                                   int* __restrict__ tki) {
  __shared__ unsigned hist[256];
  __shared__ unsigned sPrefix, sRem;
  __shared__ unsigned sGt[256], sEq[256];
  const int b = blockIdx.x;
  const int tid = threadIdx.x;
  float* p = z + (size_t)b * S_DIM;

  if (tid == 0) { sPrefix = 0u; sRem = TOPK; }
  __syncthreads();

  for (int pass = 0; pass < 4; ++pass) {
    const int shift = 24 - 8 * pass;
    const unsigned highMask = (pass == 0) ? 0u : (0xFFFFFFFFu << (shift + 8));
    hist[tid] = 0u;
    __syncthreads();
    const unsigned pref = sPrefix;
    for (int i = tid; i < S_DIM; i += 256) {
      unsigned u = fkey(p[i]);
      if ((u & highMask) == pref) atomicAdd(&hist[(u >> shift) & 255u], 1u);
    }
    __syncthreads();
    if (tid == 0) {
      unsigned rem = sRem, pfx = sPrefix;
      for (int bin = 255; bin >= 0; --bin) {
        unsigned c = hist[bin];
        if (c >= rem) { pfx |= ((unsigned)bin) << shift; break; }
        rem -= c;
      }
      sPrefix = pfx; sRem = rem;
    }
    __syncthreads();
  }

  const unsigned tau  = sPrefix;  // key of the k-th largest element
  const unsigned need = sRem;     // ties at tau to take (lowest index first)

  const int i0 = tid * 64;
  unsigned cg = 0, ce = 0;
  for (int j = 0; j < 64; ++j) {
    unsigned u = fkey(p[i0 + j]);
    if (u > tau) ++cg; else if (u == tau) ++ce;
  }
  sGt[tid] = cg; sEq[tid] = ce;
  __syncthreads();
  if (tid == 0) {            // serial exclusive prefix: deterministic, cheap
    unsigned ag = 0, ae = 0;
    for (int t = 0; t < 256; ++t) {
      unsigned g = sGt[t], e = sEq[t];
      sGt[t] = ag; sEq[t] = ae;
      ag += g; ae += e;
    }
  }
  __syncthreads();

  unsigned gtB = sGt[tid], eqB = sEq[tid];
  for (int j = 0; j < 64; ++j) {
    int i = i0 + j;
    float v = p[i];
    unsigned u = fkey(v);
    bool kept = false; unsigned pos = 0;
    if (u > tau) {
      pos = gtB + (eqB < need ? eqB : need); kept = true; ++gtB;
    } else if (u == tau) {
      if (eqB < need) { pos = gtB + eqB; kept = true; }
      ++eqB;
    }
    float zv = kept ? fmaxf(v, 0.0f) : 0.0f;
    p[i] = zv;                               // z written in place over pre
    if (kept) { tkv[b * TOPK + pos] = zv; tki[b * TOPK + pos] = i; }
  }
}

// ---------------------------------------------------------------------------
// Kernel 3: sparse decode x_hat[b,:] = sum_j val_j * W_dec[idx_j,:] + b_dec,
// plus per-row squared-error sum (fixed-order tree reduce -> deterministic).
// ---------------------------------------------------------------------------
__global__ __launch_bounds__(256) void dec_kernel(const float* __restrict__ tkv,
                                                  const int* __restrict__ tki,
                                                  const float* __restrict__ Wd,
                                                  const float* __restrict__ bdec,
                                                  const float* __restrict__ X,
                                                  float* __restrict__ xhat,
                                                  float* __restrict__ bsums) {
  __shared__ float sv[TOPK];
  __shared__ int   si[TOPK];
  __shared__ float red[256];
  const int b = blockIdx.x;
  const int tid = threadIdx.x;
  if (tid < TOPK) { sv[tid] = tkv[b * TOPK + tid]; si[tid] = tki[b * TOPK + tid]; }
  __syncthreads();

  float lsum = 0.0f;
  for (int o = tid; o < KDIM; o += 256) {
    float acc = bdec[o];
#pragma unroll 8
    for (int j = 0; j < TOPK; ++j)
      acc = fmaf(sv[j], Wd[(size_t)si[j] * KDIM + o], acc);
    float xv = X[(size_t)b * KDIM + o];
    xhat[(size_t)b * KDIM + o] = acc;
    float d = acc - xv;
    lsum = fmaf(d, d, lsum);
  }
  red[tid] = lsum;
  __syncthreads();
  for (int s = 128; s > 0; s >>= 1) {
    if (tid < s) red[tid] += red[tid + s];
    __syncthreads();
  }
  if (tid == 0) bsums[b] = red[0];
}

// ---------------------------------------------------------------------------
// Kernel 4: loss = sum(bsums) / (B*T)
// ---------------------------------------------------------------------------
__global__ __launch_bounds__(256) void fin_kernel(const float* __restrict__ bsums,
                                                  float* __restrict__ out) {
  __shared__ float red[256];
  const int tid = threadIdx.x;
  float s = 0.0f;
  for (int i = tid; i < B_ROWS; i += 256) s += bsums[i];
  red[tid] = s;
  __syncthreads();
  for (int st = 128; st > 0; st >>= 1) {
    if (tid < st) red[tid] += red[tid + st];
    __syncthreads();
  }
  if (tid == 0) out[0] = red[0] * (1.0f / 8192.0f);
}

extern "C" void kernel_launch(void* const* d_in, const int* in_sizes, int n_in,
                              void* d_out, int out_size, void* d_ws, size_t ws_size,
                              hipStream_t stream) {
  const float* x     = (const float*)d_in[0];   // (2048, 4, 768)
  const float* W_enc = (const float*)d_in[1];   // (4, 768, 16384)
  const float* b_enc = (const float*)d_in[2];   // (16384,)
  const float* W_dec = (const float*)d_in[3];   // (16384, 4, 768)
  const float* b_dec = (const float*)d_in[4];   // (4, 768)
  // d_in[5] is k == 128 (baked in as TOPK)

  float* out   = (float*)d_out;
  float* x_hat = out + 1;
  float* z     = out + 1 + (size_t)B_ROWS * KDIM;   // pre lives here, then z in place

  float* tkv   = (float*)d_ws;                          // 2048*128 f32
  int*   tki   = (int*)((float*)d_ws + B_ROWS * TOPK);  // 2048*128 i32
  float* bsums = (float*)d_ws + 2 * B_ROWS * TOPK;      // 2048 f32

  enc_kernel<<<dim3(B_ROWS / 128, S_DIM / 128), 256, 0, stream>>>(x, W_enc, b_enc, z);
  topk_kernel<<<B_ROWS, 256, 0, stream>>>(z, tkv, tki);
  dec_kernel<<<B_ROWS, 256, 0, stream>>>(tkv, tki, W_dec, b_dec, x, x_hat, bsums);
  fin_kernel<<<1, 256, 0, stream>>>(bsums, out);
}
